// TransformerBlocks_50921132261733
// MI455X (gfx1250) — matrix operand order repaired and checked by reading, not compiled, not run on hardware
//
#include <hip/hip_runtime.h>
#include <hip/hip_bf16.h>
#include <stdint.h>

// ---------------------------------------------------------------------------
// MI455X (gfx1250) transformer block.
// bf16 operands + fp32 accumulation on v_wmma_f32_16x16x32_bf16 for all GEMMs
// and both attention contractions; fp32 for softmax stats / GroupNorm /
// residuals. All fragment loads are b128 via WMMA-friendly packed layouts:
//   A operand : row-major [M,K], K % 8 == 0  -> 2x uint4 per fragment
//   B operand : packed groups Bp[k/16][col][16] -> 2x uint4 per fragment
// K is padded to a multiple of 32 (1365 -> 1376) so the GEMM has no tail.
// ---------------------------------------------------------------------------

typedef __attribute__((ext_vector_type(16))) __bf16 v16bf;
typedef __attribute__((ext_vector_type(8)))  float  v8f;

#define DEVINL static __device__ __forceinline__

DEVINL unsigned short f2bf(float f) {
  unsigned int u = __builtin_bit_cast(unsigned int, f);
  unsigned int r = u + 0x7fffu + ((u >> 16) & 1u);   // round-to-nearest-even
  return (unsigned short)(r >> 16);
}

struct Q2 { uint4 a, b; };
DEVINL v16bf ld_frag(const unsigned short* p0, const unsigned short* p1) {
  Q2 t;
  t.a = *(const uint4*)p0;   // 8 bf16
  t.b = *(const uint4*)p1;   // 8 bf16
  return __builtin_bit_cast(v16bf, t);
}

DEVINL v8f wmma_bf16(v16bf a, v16bf b, v8f c) {
  return __builtin_amdgcn_wmma_f32_16x16x32_bf16(false, a, false, b, (short)0,
                                                 c, false, false);
}

// ---------------------------------------------------------------------------
// fp32 -> bf16, row-major (A operands / weights)
// ---------------------------------------------------------------------------
__global__ void cvt_f32_bf16(const float* __restrict__ in,
                             unsigned short* __restrict__ out, size_t n) {
  size_t i = (size_t)blockIdx.x * blockDim.x + threadIdx.x;
  if (i < n) out[i] = f2bf(in[i]);
}

// fp32 [M,K] -> bf16 [M,Kp] row-major, zero-padded columns K..Kp
__global__ void cvt_a_pad(const float* __restrict__ in,
                          unsigned short* __restrict__ out,
                          int M, int K, int Kp) {
  size_t i = (size_t)blockIdx.x * blockDim.x + threadIdx.x;
  size_t total = (size_t)M * Kp;
  if (i >= total) return;
  int m = (int)(i / Kp), k = (int)(i % Kp);
  out[i] = (k < K) ? f2bf(in[(size_t)m * K + k]) : (unsigned short)0;
}

// fp32 [B][R,N] -> packed bf16 B-operand groups: out[b][(k>>4)*N + n][16] pos k&15
__global__ void cvt_bpack(const float* __restrict__ in,
                          unsigned short* __restrict__ out,
                          int R, int Nn, size_t total) {
  size_t i = (size_t)blockIdx.x * blockDim.x + threadIdx.x;
  if (i >= total) return;
  size_t RN = (size_t)R * Nn;
  size_t b = i / RN, rem = i % RN;
  int k = (int)(rem / Nn), n = (int)(rem % Nn);
  out[b * RN + (((size_t)(k >> 4) * Nn + n) << 4) + (k & 15)] = f2bf(in[i]);
}

// v slice of qkv -> packed PV B-operand per (b,h): keys are the K dim.
// out[bh][(n>>4)*64 + d][16] pos n&15
__global__ void cvt_v_pack(const float* __restrict__ qkv,
                           unsigned short* __restrict__ v,
                           int Hh, int Dd, int Nn, size_t total) {
  size_t i = (size_t)blockIdx.x * blockDim.x + threadIdx.x;
  if (i >= total) return;                    // total = B*HD*N
  size_t HDN = (size_t)Hh * Dd * Nn;
  size_t b = i / HDN, rem = i % HDN;
  int c = (int)(rem / Nn), n = (int)(rem % Nn);
  int h = c / Dd, dd = c % Dd;
  float val = qkv[b * 3 * HDN + 2 * HDN + rem];
  size_t bh = b * Hh + h;
  v[bh * (size_t)Dd * Nn + (((size_t)(n >> 4) * Dd + dd) << 4) + (n & 15)] = f2bf(val);
}

// zero the pad rows [k_lo,k_hi) of a packed B-operand buffer
__global__ void zero_pad_rows(unsigned short* __restrict__ out, size_t bstride,
                              int Nn, int k_lo, int k_hi, size_t total) {
  size_t i = (size_t)blockIdx.x * blockDim.x + threadIdx.x;
  if (i >= total) return;                    // total = B*(k_hi-k_lo)*N
  int span = k_hi - k_lo;
  size_t PN = (size_t)span * Nn;
  size_t b = i / PN, rem = i % PN;
  int k = k_lo + (int)(rem / Nn), n = (int)(rem % Nn);
  out[b * bstride + (((size_t)(k >> 4) * Nn + n) << 4) + (k & 15)] = 0;
}

// ---------------------------------------------------------------------------
// GEMM: Out[b] = A[MxK] * B[b][KxN] (+bias) (+res). One 32x32 tile per wave
// (2x2 WMMA subtiles), K stepped by 32, all operand loads are b128.
// ---------------------------------------------------------------------------
__global__ __launch_bounds__(128)
void gemm_bf16_wmma(const unsigned short* __restrict__ A,   // [M,K] row-major
                    const unsigned short* __restrict__ Bp,  // packed [K/16][Nc][16]
                    float* __restrict__ Out,
                    const float* __restrict__ bias,
                    const float* __restrict__ res,
                    int M, int K, int Nc) {
  const int lane = threadIdx.x & 31;
  const int wid  = blockIdx.x * (blockDim.x >> 5) + (threadIdx.x >> 5);
  const int nt32 = Nc >> 5;
  const int mt32 = (M + 31) >> 5;
  if (wid >= mt32 * nt32) return;            // wave-uniform
  const int mt = wid / nt32, nt = wid % nt32;
  const int b = blockIdx.z;

  const unsigned short* Bb = Bp + (size_t)b * K * Nc;
  float* Ob = Out + (size_t)b * M * Nc;
  const float* Rb = res ? res + (size_t)b * M * Nc : nullptr;

  const int hh = lane >> 4, l16 = lane & 15;
  int row0 = mt * 32 + l16;      if (row0 >= M) row0 = M - 1;   // clamp (OOB rows unstored)
  int row1 = mt * 32 + 16 + l16; if (row1 >= M) row1 = M - 1;
  const int col0 = nt * 32 + l16;

  const unsigned short* A0 = A + (size_t)row0 * K + 16 * hh;
  const unsigned short* A1 = A + (size_t)row1 * K + 16 * hh;

  v8f acc[2][2] = {{v8f{}, v8f{}}, {v8f{}, v8f{}}};
  for (int k0 = 0; k0 < K; k0 += 32) {
    __builtin_prefetch(A0 + k0 + 32, 0, 1);
    v16bf a0 = ld_frag(A0 + k0, A0 + k0 + 8);
    v16bf a1 = ld_frag(A1 + k0, A1 + k0 + 8);
    const unsigned short* Bg = Bb + ((size_t)((k0 >> 4) + hh) * Nc) * 16;
    const unsigned short* B0 = Bg + (size_t)col0 * 16;
    const unsigned short* B1 = B0 + 16 * 16;       // col0 + 16
    v16bf b0 = ld_frag(B0, B0 + 8);
    v16bf b1 = ld_frag(B1, B1 + 8);
    acc[0][0] = wmma_bf16(a0, b0, acc[0][0]);
    acc[0][1] = wmma_bf16(a0, b1, acc[0][1]);
    acc[1][0] = wmma_bf16(a1, b0, acc[1][0]);
    acc[1][1] = wmma_bf16(a1, b1, acc[1][1]);
  }
  // C/D layout: col = 16*ns + l16, rows = 16*ms + 8*hh + r
#pragma unroll
  for (int ms = 0; ms < 2; ++ms)
#pragma unroll
    for (int r = 0; r < 8; ++r) {
      int m = mt * 32 + ms * 16 + hh * 8 + r;
      if (m < M) {
        float bia = bias ? bias[m] : 0.f;
#pragma unroll
        for (int ns = 0; ns < 2; ++ns) {
          size_t idx = (size_t)m * Nc + nt * 32 + ns * 16 + l16;
          float vv = acc[ms][ns][r] + bia;
          if (Rb) vv += Rb[idx];
          Ob[idx] = vv;
        }
      }
    }
}

// ---------------------------------------------------------------------------
// q/k: l2norm over D + per-dim scale.
//   q -> bf16 [B,H,N,D] row-major (A operand), k -> packed B-operand per (b,h)
// ---------------------------------------------------------------------------
__global__ void qk_norm_kernel(const float* __restrict__ qkv,
                               const float* __restrict__ qs,
                               const float* __restrict__ ks,
                               unsigned short* __restrict__ qt,
                               unsigned short* __restrict__ kb,
                               int Bn, int Hh, int Dd, int Nn) {
  int i = blockIdx.x * blockDim.x + threadIdx.x;
  int total = Bn * Hh * Nn;
  if (i >= total) return;
  int n = i % Nn, bh = i / Nn;
  int h = bh % Hh, b = bh / Hh;
  size_t HD = (size_t)Hh * Dd;
  const float* qp = qkv + ((size_t)b * 3 * HD + (size_t)h * Dd) * Nn + n;
  const float* kp = qp + HD * Nn;
  float sq = 0.f, sk = 0.f;
  for (int d = 0; d < Dd; ++d) {
    float q = qp[(size_t)d * Nn], k = kp[(size_t)d * Nn];
    sq += q * q; sk += k * k;
  }
  float rq = rsqrtf(sq + 1e-12f), rk = rsqrtf(sk + 1e-12f);
  unsigned short* qo = qt + ((size_t)bh * Nn + n) * Dd;
  unsigned short* ko = kb + (size_t)bh * Dd * Nn;
  for (int d = 0; d < Dd; ++d) {
    qo[d] = f2bf(qp[(size_t)d * Nn] * rq * qs[d]);
    ko[(((size_t)(d >> 4) * Nn + n) << 4) + (d & 15)] = f2bf(kp[(size_t)d * Nn] * rk * ks[d]);
  }
}

// ---------------------------------------------------------------------------
// Flash attention: one wave per (b,h,16-query tile); 32-key blocks.
// Scores via 4 WMMAs, online softmax with 16-lane shfl_xor butterflies,
// P relaid C->A through LDS (b128 reload), PV via 4 WMMAs.
// Output written directly in packed B-operand layout for the out-projection.
// ---------------------------------------------------------------------------
__global__ __launch_bounds__(128)
void flash_attn(const unsigned short* __restrict__ qt,  // [B,H,N,64] row-major
                const unsigned short* __restrict__ kbm, // packed per (b,h): [4][N][16]
                const unsigned short* __restrict__ vbm, // packed per (b,h): [N/16][64][16]
                unsigned short* __restrict__ ao,        // packed per b: [HD/16][N][16]
                int Bn, int Hh, int Nn) {
  __shared__ __align__(16) unsigned short smem[4][16 * 32];
  const int lane = threadIdx.x & 31;
  const int slot = threadIdx.x >> 5;
  const int wid = blockIdx.x * 4 + slot;
  const int qtiles = Nn >> 4;
  const int qtile = wid % qtiles;
  const int bh = wid / qtiles;
  if (bh >= Bn * Hh) return;
  const int hh = lane >> 4, l16 = lane & 15;
  const int q0 = qtile * 16;

  const unsigned short* Q  = qt  + (size_t)bh * Nn * 64;
  const unsigned short* Km = kbm + (size_t)bh * 64 * Nn;
  const unsigned short* Vm = vbm + (size_t)bh * 64 * Nn;

  // Q A-fragments for the two D chunks (b128 loads)
  v16bf qa[2];
  {
    const unsigned short* Qr = Q + (size_t)(q0 + l16) * 64 + 16 * hh;
#pragma unroll
    for (int c = 0; c < 2; ++c) qa[c] = ld_frag(Qr + c * 32, Qr + c * 32 + 8);
  }

  v8f o[4] = {v8f{}, v8f{}, v8f{}, v8f{}};
  float mrow[8], lrow[8];
#pragma unroll
  for (int r = 0; r < 8; ++r) { mrow[r] = -3.0e38f; lrow[r] = 0.f; }

  unsigned short* myLds = smem[slot];

  for (int j0 = 0; j0 < Nn; j0 += 32) {
    v8f s0 = {}, s1 = {};
#pragma unroll
    for (int c = 0; c < 2; ++c) {                 // D chunk (K dim of QK^T)
      // packed K: group (2c+hh), columns j0+l16 / j0+16+l16
      const unsigned short* Kg = Km + ((size_t)(2 * c + hh) * Nn) * 16;
      const unsigned short* K0 = Kg + (size_t)(j0 + l16) * 16;
      const unsigned short* K1 = Kg + (size_t)(j0 + 16 + l16) * 16;
      v16bf k0f = ld_frag(K0, K0 + 8);
      v16bf k1f = ld_frag(K1, K1 + 8);
      s0 = wmma_bf16(qa[c], k0f, s0);
      s1 = wmma_bf16(qa[c], k1f, s1);
    }
    // online softmax over this 32-key block (row = query, 16 cols across lanes)
    float p0[8], p1[8];
#pragma unroll
    for (int r = 0; r < 8; ++r) {
      float a = s0[r] * 8.0f, b2 = s1[r] * 8.0f;   // SCALE = 8
      float rm = fmaxf(a, b2);
      rm = fmaxf(rm, __shfl_xor(rm, 1, 32));
      rm = fmaxf(rm, __shfl_xor(rm, 2, 32));
      rm = fmaxf(rm, __shfl_xor(rm, 4, 32));
      rm = fmaxf(rm, __shfl_xor(rm, 8, 32));
      float nm = fmaxf(mrow[r], rm);
      float alpha = __expf(mrow[r] - nm);
      float e0 = __expf(a - nm), e1 = __expf(b2 - nm);
      float rs = e0 + e1;
      rs += __shfl_xor(rs, 1, 32);
      rs += __shfl_xor(rs, 2, 32);
      rs += __shfl_xor(rs, 4, 32);
      rs += __shfl_xor(rs, 8, 32);
      lrow[r] = lrow[r] * alpha + rs;
      mrow[r] = nm;
      p0[r] = e0; p1[r] = e1;
#pragma unroll
      for (int dt = 0; dt < 4; ++dt) o[dt][r] *= alpha;
    }
    // P: C-layout regs -> LDS (row-major 16x32 bf16) -> A-layout (same-wave DS order)
#pragma unroll
    for (int r = 0; r < 8; ++r) {
      int row = hh * 8 + r;
      myLds[row * 32 + l16]      = f2bf(p0[r]);
      myLds[row * 32 + 16 + l16] = f2bf(p1[r]);
    }
    const unsigned short* Pr = myLds + l16 * 32 + 16 * hh;
    v16bf pa = ld_frag(Pr, Pr + 8);               // 2x ds_load_b128
    // PV: packed V, group (j0>>4)+hh, column = d
#pragma unroll
    for (int dt = 0; dt < 4; ++dt) {
      const unsigned short* Vp =
          Vm + (((size_t)((j0 >> 4) + hh) * 64 + dt * 16 + l16)) * 16;
      v16bf vf = ld_frag(Vp, Vp + 8);
      o[dt] = wmma_bf16(pa, vf, o[dt]);
    }
  }
  // normalize; emit packed B-operand layout: per b, group (h*4+dt), col n, pos l16
  int b = bh / Hh, h = bh % Hh;
  unsigned short* aob = ao + (size_t)b * Hh * 64 * Nn;
#pragma unroll
  for (int r = 0; r < 8; ++r) {
    float inv = 1.0f / lrow[r];
    int n = q0 + hh * 8 + r;
#pragma unroll
    for (int dt = 0; dt < 4; ++dt)
      aob[(((size_t)(h * 4 + dt) * Nn + n) << 4) + l16] = f2bf(o[dt][r] * inv);
  }
}

// ---------------------------------------------------------------------------
// GroupNorm(1,C): stats over the whole [C,N] plane per sample.
// ---------------------------------------------------------------------------
__global__ void gn_reduce(const float* __restrict__ in, long P,
                          float2* __restrict__ partials) {
  int b = blockIdx.y;
  const float* p = in + (size_t)b * P;
  float s = 0.f, ss = 0.f;
  for (long i = (long)blockIdx.x * blockDim.x + threadIdx.x; i < P;
       i += (long)gridDim.x * blockDim.x) {
    float v = p[i];
    s += v; ss += v * v;
  }
  for (int m = 16; m >= 1; m >>= 1) { s += __shfl_xor(s, m, 32); ss += __shfl_xor(ss, m, 32); }
  __shared__ float sh[2][8];
  int wv = threadIdx.x >> 5;
  if ((threadIdx.x & 31) == 0) { sh[0][wv] = s; sh[1][wv] = ss; }
  __syncthreads();
  if (threadIdx.x == 0) {
    float S = 0.f, SS = 0.f;
    for (int w = 0; w < (int)(blockDim.x >> 5); ++w) { S += sh[0][w]; SS += sh[1][w]; }
    partials[(size_t)b * gridDim.x + blockIdx.x] = make_float2(S, SS);
  }
}

__global__ void gn_finalize(const float2* __restrict__ partials, int nblk, long P,
                            float2* __restrict__ stats) {
  int b = blockIdx.x;
  float s = 0.f, ss = 0.f;
  for (int i = threadIdx.x; i < nblk; i += blockDim.x) {
    float2 v = partials[(size_t)b * nblk + i];
    s += v.x; ss += v.y;
  }
  for (int m = 16; m >= 1; m >>= 1) { s += __shfl_xor(s, m, 32); ss += __shfl_xor(ss, m, 32); }
  __shared__ float sh[2][8];
  int wv = threadIdx.x >> 5;
  if ((threadIdx.x & 31) == 0) { sh[0][wv] = s; sh[1][wv] = ss; }
  __syncthreads();
  if (threadIdx.x == 0) {
    float S = 0.f, SS = 0.f;
    for (int w = 0; w < (int)(blockDim.x >> 5); ++w) { S += sh[0][w]; SS += sh[1][w]; }
    float mean = S / (float)P;
    float var = SS / (float)P - mean * mean;
    stats[b] = make_float2(mean, rsqrtf(var + 1e-5f));
  }
}

// Apply GN; optionally emit fp32 plane and/or packed bf16 B-operand
__global__ void gn_apply(const float* __restrict__ in, const float* __restrict__ g,
                         const float* __restrict__ bta, const float2* __restrict__ stats,
                         float* __restrict__ outf, unsigned short* __restrict__ outb,
                         size_t bstride_pk, int Nn, long P) {
  long i = (long)blockIdx.x * blockDim.x + threadIdx.x;
  if (i >= P) return;
  int b = blockIdx.y;
  int c = (int)(i / Nn), n = (int)(i % Nn);
  float2 st = stats[b];
  float v = (in[(size_t)b * P + i] - st.x) * st.y * g[c] + bta[c];
  if (outf) outf[(size_t)b * P + i] = v;
  if (outb) outb[(size_t)b * bstride_pk + (((size_t)(c >> 4) * Nn + n) << 4) + (c & 15)] = f2bf(v);
}

// ---------------------------------------------------------------------------
// SwiGLU: out = gate * silu(a)
// ---------------------------------------------------------------------------
__global__ void swiglu_kernel(const float* __restrict__ h, float* __restrict__ out,
                              int inner, int Nn) {
  long P = (long)inner * Nn;
  long i = (long)blockIdx.x * blockDim.x + threadIdx.x;
  if (i >= P) return;
  int b = blockIdx.y;
  float a = h[(size_t)b * 2 * P + i];
  float gate = h[(size_t)b * 2 * P + P + i];
  out[(size_t)b * P + i] = gate * (a / (1.0f + __expf(-a)));
}

// ---------------------------------------------------------------------------
extern "C" void kernel_launch(void* const* d_in, const int* in_sizes, int n_in,
                              void* d_out, int out_size, void* d_ws, size_t ws_size,
                              hipStream_t stream) {
  const int B = 2, C = 512, N = 2048, H = 16, D = 64, HD = 1024;
  const int INNER = 1365, I2 = 2730, THD3 = 3072;
  const int KP2 = 1376;  // INNER padded to multiple of 32
  (void)in_sizes; (void)n_in; (void)out_size; (void)ws_size;

  const float* x       = (const float*)d_in[0];
  const float* qkv_w   = (const float*)d_in[1];
  const float* q_scale = (const float*)d_in[2];
  const float* k_scale = (const float*)d_in[3];
  const float* out_w   = (const float*)d_in[4];
  const float* out_b   = (const float*)d_in[5];
  const float* ff_g1   = (const float*)d_in[6];
  const float* ff_b1   = (const float*)d_in[7];
  const float* ff_w1   = (const float*)d_in[8];
  const float* ff_g2   = (const float*)d_in[9];
  const float* ff_b2   = (const float*)d_in[10];
  const float* ff_w2   = (const float*)d_in[11];
  const float* norm_g  = (const float*)d_in[12];
  const float* norm_b  = (const float*)d_in[13];
  float* out = (float*)d_out;

  char* p = (char*)d_ws;
  auto carve = [&](size_t bytes) -> char* {
    char* r = p;
    p += (bytes + 255) & ~(size_t)255;
    return r;
  };
  unsigned short* x_pk  = (unsigned short*)carve((size_t)B * C * N * 2);      // B-op packed
  unsigned short* wq_bf = (unsigned short*)carve((size_t)THD3 * C * 2);       // A row-major
  unsigned short* wo_bf = (unsigned short*)carve((size_t)C * HD * 2);
  unsigned short* w1_bf = (unsigned short*)carve((size_t)I2 * C * 2);
  unsigned short* w2_bf = (unsigned short*)carve((size_t)C * KP2 * 2);        // A padded
  float*          qkv_f = (float*)carve((size_t)B * THD3 * N * 4);
  unsigned short* qt_bf = (unsigned short*)carve((size_t)B * H * N * D * 2);  // A row-major
  unsigned short* k_pk  = (unsigned short*)carve((size_t)B * H * D * N * 2);  // packed
  unsigned short* v_pk  = (unsigned short*)carve((size_t)B * H * D * N * 2);  // packed
  unsigned short* ao_pk = (unsigned short*)carve((size_t)B * HD * N * 2);     // packed
  float*          x1_f  = (float*)carve((size_t)B * C * N * 4);
  unsigned short* hn_pk = (unsigned short*)carve((size_t)B * C * N * 2);      // packed
  float*          ff1_f = (float*)carve((size_t)B * I2 * N * 4);
  float*          sw_f  = (float*)carve((size_t)B * INNER * N * 4);
  unsigned short* sn_pk = (unsigned short*)carve((size_t)B * KP2 * N * 2);    // packed+pad
  float*          x2_f  = (float*)carve((size_t)B * C * N * 4);
  float2*         parts = (float2*)carve((size_t)B * 512 * sizeof(float2));
  float2*         stats = (float2*)carve((size_t)B * sizeof(float2));

  auto cdiv = [](long a, long b) -> int { return (int)((a + b - 1) / b); };
  const int NB = 512;  // GN partial blocks

  // -- operand staging to bf16 --
  cvt_bpack<<<cdiv((long)B * C * N, 256), 256, 0, stream>>>(x, x_pk, C, N, (size_t)B * C * N);
  cvt_f32_bf16<<<cdiv((long)THD3 * C, 256), 256, 0, stream>>>(qkv_w, wq_bf, (size_t)THD3 * C);
  cvt_f32_bf16<<<cdiv((long)C * HD, 256), 256, 0, stream>>>(out_w, wo_bf, (size_t)C * HD);
  cvt_f32_bf16<<<cdiv((long)I2 * C, 256), 256, 0, stream>>>(ff_w1, w1_bf, (size_t)I2 * C);
  cvt_a_pad<<<cdiv((long)C * KP2, 256), 256, 0, stream>>>(ff_w2, w2_bf, C, INNER, KP2);

  // -- QKV projection --
  {
    int waves = ((THD3 + 31) / 32) * (N / 32);
    gemm_bf16_wmma<<<dim3(cdiv(waves, 4), 1, B), 128, 0, stream>>>(
        wq_bf, x_pk, qkv_f, nullptr, nullptr, THD3, C, N);
  }
  // -- q/k l2norm + scale; v pack --
  qk_norm_kernel<<<cdiv((long)B * H * N, 256), 256, 0, stream>>>(
      qkv_f, q_scale, k_scale, qt_bf, k_pk, B, H, D, N);
  cvt_v_pack<<<cdiv((long)B * HD * N, 256), 256, 0, stream>>>(
      qkv_f, v_pk, H, D, N, (size_t)B * HD * N);
  // -- attention --
  flash_attn<<<dim3((B * H * (N / 16)) / 4), 128, 0, stream>>>(
      qt_bf, k_pk, v_pk, ao_pk, B, H, N);
  // -- out projection + bias + residual --
  {
    int waves = ((C + 31) / 32) * (N / 32);
    gemm_bf16_wmma<<<dim3(cdiv(waves, 4), 1, B), 128, 0, stream>>>(
        wo_bf, ao_pk, x1_f, out_b, x, C, HD, N);
  }
  // -- GN1 -> packed bf16 --
  gn_reduce<<<dim3(NB, B), 256, 0, stream>>>(x1_f, (long)C * N, parts);
  gn_finalize<<<B, 256, 0, stream>>>(parts, NB, (long)C * N, stats);
  gn_apply<<<dim3(cdiv((long)C * N, 256), B), 256, 0, stream>>>(
      x1_f, ff_g1, ff_b1, stats, nullptr, hn_pk, (size_t)C * N, N, (long)C * N);
  // -- FF1 --
  {
    int waves = ((I2 + 31) / 32) * (N / 32);
    gemm_bf16_wmma<<<dim3(cdiv(waves, 4), 1, B), 128, 0, stream>>>(
        w1_bf, hn_pk, ff1_f, nullptr, nullptr, I2, C, N);
  }
  // -- SwiGLU --
  swiglu_kernel<<<dim3(cdiv((long)INNER * N, 256), B), 256, 0, stream>>>(ff1_f, sw_f, INNER, N);
  // -- GN2 -> packed bf16 (+ zero pad rows INNER..KP2) --
  gn_reduce<<<dim3(NB, B), 256, 0, stream>>>(sw_f, (long)INNER * N, parts);
  gn_finalize<<<B, 256, 0, stream>>>(parts, NB, (long)INNER * N, stats);
  gn_apply<<<dim3(cdiv((long)INNER * N, 256), B), 256, 0, stream>>>(
      sw_f, ff_g2, ff_b2, stats, nullptr, sn_pk, (size_t)KP2 * N, N, (long)INNER * N);
  zero_pad_rows<<<cdiv((long)B * (KP2 - INNER) * N, 256), 256, 0, stream>>>(
      sn_pk, (size_t)KP2 * N, N, INNER, KP2, (size_t)B * (KP2 - INNER) * N);
  // -- FF2 (K padded, no tail) + residual --
  {
    int waves = ((C + 31) / 32) * (N / 32);
    gemm_bf16_wmma<<<dim3(cdiv(waves, 4), 1, B), 128, 0, stream>>>(
        w2_bf, sn_pk, x2_f, nullptr, x1_f, C, KP2, N);
  }
  // -- final GroupNorm -> d_out (fp32) --
  gn_reduce<<<dim3(NB, B), 256, 0, stream>>>(x2_f, (long)C * N, parts);
  gn_finalize<<<B, 256, 0, stream>>>(parts, NB, (long)C * N, stats);
  gn_apply<<<dim3(cdiv((long)C * N, 256), B), 256, 0, stream>>>(
      x2_f, norm_g, norm_b, stats, out, nullptr, 0, N, (long)C * N);
}
